// OctreeHead_38379827757293
// MI455X (gfx1250) — compile-verified, matrix operand order checked by reading, not run on hardware
//
#include <hip/hip_runtime.h>
#include <hip/hip_bf16.h>
#include <math.h>

typedef __attribute__((ext_vector_type(16))) _Float16 v16h;
typedef __attribute__((ext_vector_type(8)))  _Float16 half8;
typedef __attribute__((ext_vector_type(8)))  float    v8f;

#define CIN  128
#define CMID 128
#define COUT 32
#define KNB  27
#define LDA  136   // 128 + 8 halves pad -> 272B row stride (16B multiple), no bank conflicts

#define H8Z {(_Float16)0,(_Float16)0,(_Float16)0,(_Float16)0,(_Float16)0,(_Float16)0,(_Float16)0,(_Float16)0}

// ---------------------------------------------------------------------------
// prep: x (fp32) -> f16 ; W1 [27,128,128] -> W1t [27,128(out),128(in)] f16 ;
//       W2 [27,128,32]  -> W2t [27,32(out),128(in)]  f16
// ---------------------------------------------------------------------------
__global__ void prep_convert(const float* __restrict__ x,
                             const float* __restrict__ W1,
                             const float* __restrict__ W2,
                             _Float16* __restrict__ xb,
                             _Float16* __restrict__ W1t,
                             _Float16* __restrict__ W2t,
                             int N) {
    long long i = (long long)blockIdx.x * blockDim.x + threadIdx.x;
    long long nx = (long long)N * CIN;
    const long long nw1 = (long long)KNB * CIN * CMID;   // 442368
    const long long nw2 = (long long)KNB * CMID * COUT;  // 110592
    if (i < nx) { xb[i] = (_Float16)x[i]; return; }
    i -= nx;
    if (i < nw1) {
        int k = (int)(i >> 14);       // /16384
        int r = (int)(i & 16383);
        int n = r >> 7;               // out channel
        int c = r & 127;              // in channel
        W1t[i] = (_Float16)W1[(long long)k * 16384 + (long long)c * 128 + n];
        return;
    }
    i -= nw1;
    if (i < nw2) {
        int k = (int)(i >> 12);       // /4096
        int r = (int)(i & 4095);
        int n = r >> 7;               // out channel (0..31)
        int c = r & 127;              // in channel
        W2t[i] = (_Float16)W2[(long long)k * 4096 + (long long)c * 32 + n];
    }
}

// ---------------------------------------------------------------------------
// 16x32 f16 WMMA fragment from an LDS row: two contiguous 16B chunks
// lane l15 selects row/col, kh selects K-half group (ISA 7.12.2 layout)
// ---------------------------------------------------------------------------
__device__ __forceinline__ v16h load_frag(const _Float16* rowp, int kc, int kh) {
    union { v16h v; half8 h[2]; } u;
    u.h[0] = *(const half8*)(rowp + kc + kh * 8);
    u.h[1] = *(const half8*)(rowp + kc + 16 + kh * 8);
    return u.v;
}

// ---------------------------------------------------------------------------
// conv1: y1[N,128] = gather(xb, neigh) @ W1 + b1 ; also per-group sum/sumsq
// block: 256 thr = 8 waves; tile 128 rows x 128 cols; B staged once per k,
// A staged in two 64-row halves reusing one LDS buffer. kc loop is manually
// software-pipelined so ds_loads of kc+1 issue before wmmas of kc.
// ---------------------------------------------------------------------------
__launch_bounds__(256)
__global__ void conv1_wmma(const _Float16* __restrict__ xb,
                           const long long* __restrict__ neigh,
                           const _Float16* __restrict__ W1t,
                           const float* __restrict__ b1,
                           float* __restrict__ y1,
                           float* __restrict__ gsum,   // [32 groups][sum,sumsq]
                           int N) {
    __shared__ __align__(16) _Float16 As[64][LDA];
    __shared__ __align__(16) _Float16 Bs[128][LDA];
    __shared__ float gred[64];

    const int tid  = threadIdx.x;
    const int lane = tid & 31;
    const int wave = tid >> 5;
    const int l15  = lane & 15;
    const int kh   = lane >> 4;     // 0/1
    const int wm   = wave >> 1;     // 0..3 : row tile within half
    const int wn   = wave & 1;      // 0..1 : col half
    const int rowBase0 = blockIdx.x * 128;

    // staging roles: 4 threads per A row (32 halves each), 2 per B row
    const int arow_s = tid >> 2;            // 0..63
    const int aoff_s = (tid & 3) * 32;      // 0,32,64,96
    const int brow_s = tid >> 1;            // 0..127
    const int boff_s = (tid & 1) * 64;      // 0,64

    const v8f vzero = {0.f,0.f,0.f,0.f,0.f,0.f,0.f,0.f};
    v8f acc[8];
#pragma unroll
    for (int i = 0; i < 8; ++i) acc[i] = vzero;

    for (int k = 0; k < KNB; ++k) {
        const _Float16* wk = W1t + (long long)k * (CIN * CMID);
        if (k + 1 < KNB)   // warm WGP$/L2 with next weight slice (32KB = 256 lines)
            __builtin_prefetch(W1t + (long long)(k + 1) * (CIN * CMID) + (long long)tid * 64, 0, 0);

        __syncthreads();   // everyone done reading As(half1,k-1) / Bs(k-1)
        // stage B once per k: 128x128 halves
#pragma unroll
        for (int i = 0; i < 8; ++i)
            *(half8*)(&Bs[brow_s][boff_s + i * 8]) =
                *(const half8*)(wk + brow_s * CIN + boff_s + i * 8);

#pragma unroll
        for (int h = 0; h < 2; ++h) {
            if (h == 1) __syncthreads();   // done reading As(half0)
            // stage A: 64 rows of this half (one neigh read per thread)
            {
                int grow = rowBase0 + h * 64 + arow_s;
                long long src = -1;
                if (grow < N) src = neigh[(long long)grow * KNB + k];
#pragma unroll
                for (int i = 0; i < 4; ++i) {
                    half8 v = H8Z;
                    if (src >= 0) v = *(const half8*)(xb + src * CIN + aoff_s + i * 8);
                    *(half8*)(&As[arow_s][aoff_s + i * 8]) = v;
                }
            }
            __syncthreads();

            const _Float16* arow = &As[wm * 16 + l15][0];
            const _Float16* b0   = &Bs[wn * 64 + l15][0];

            // software pipeline: preload kc=0 fragments, then issue kc+1 loads
            // before consuming kc's fragments in the wmmas.
            v16h af  = load_frag(arow, 0, kh);
            v16h bf0 = load_frag(b0,            0, kh);
            v16h bf1 = load_frag(b0 + 16 * LDA, 0, kh);
            v16h bf2 = load_frag(b0 + 32 * LDA, 0, kh);
            v16h bf3 = load_frag(b0 + 48 * LDA, 0, kh);
#pragma unroll
            for (int kc = 0; kc < 128; kc += 32) {
                v16h caf = af, cb0 = bf0, cb1 = bf1, cb2 = bf2, cb3 = bf3;
                if (kc < 96) {
                    af  = load_frag(arow, kc + 32, kh);
                    bf0 = load_frag(b0,            kc + 32, kh);
                    bf1 = load_frag(b0 + 16 * LDA, kc + 32, kh);
                    bf2 = load_frag(b0 + 32 * LDA, kc + 32, kh);
                    bf3 = load_frag(b0 + 48 * LDA, kc + 32, kh);
                }
                acc[h*4+0] = __builtin_amdgcn_wmma_f32_16x16x32_f16(false, caf, false, cb0, (short)0, acc[h*4+0], false, false);
                acc[h*4+1] = __builtin_amdgcn_wmma_f32_16x16x32_f16(false, caf, false, cb1, (short)0, acc[h*4+1], false, false);
                acc[h*4+2] = __builtin_amdgcn_wmma_f32_16x16x32_f16(false, caf, false, cb2, (short)0, acc[h*4+2], false, false);
                acc[h*4+3] = __builtin_amdgcn_wmma_f32_16x16x32_f16(false, caf, false, cb3, (short)0, acc[h*4+3], false, false);
            }
        }
    }

    // epilogue: bias + store + per-group partial sums
    if (tid < 64) gred[tid] = 0.f;
    __syncthreads();

#pragma unroll
    for (int h = 0; h < 2; ++h) {
#pragma unroll
        for (int nt = 0; nt < 4; ++nt) {
            int col = wn * 64 + nt * 16 + l15;
            float bias = b1[col];
            float s = 0.f, s2 = 0.f;
#pragma unroll
            for (int r = 0; r < 8; ++r) {
                int grow = rowBase0 + h * 64 + wm * 16 + kh * 8 + r;  // vgpr r -> M = kh*8+r
                float v = acc[h*4+nt][r] + bias;
                if (grow < N) {
                    y1[(long long)grow * CMID + col] = v;
                    s  += v;
                    s2 += v * v;
                }
            }
            int g = col >> 2;   // 4 channels per group
            atomicAdd(&gred[2 * g],     s);
            atomicAdd(&gred[2 * g + 1], s2);
        }
    }
    __syncthreads();
    if (tid < 64) atomicAdd(&gsum[tid], gred[tid]);
}

// ---------------------------------------------------------------------------
// fold GN stats into per-channel scale/shift
// ---------------------------------------------------------------------------
__global__ void gn_stats(const float* __restrict__ gsum,
                         const float* __restrict__ gn_w,
                         const float* __restrict__ gn_b,
                         float* __restrict__ ss, int N) {
    int c = threadIdx.x;
    if (c >= CMID) return;
    int g = c >> 2;
    float cnt  = (float)N * 4.0f;
    float mean = gsum[2 * g] / cnt;
    float var  = gsum[2 * g + 1] / cnt - mean * mean;
    float rstd = rsqrtf(var + 1e-5f);
    float scale = gn_w[c] * rstd;
    float shift = gn_b[c] - mean * scale;
    ss[2 * c]     = scale;
    ss[2 * c + 1] = shift;
}

// ---------------------------------------------------------------------------
// normalize + SiLU + convert to f16
// ---------------------------------------------------------------------------
__global__ void norm_silu(const float* __restrict__ y1,
                          const float* __restrict__ ss,
                          _Float16* __restrict__ x2, long long total) {
    long long i = (long long)blockIdx.x * blockDim.x + threadIdx.x;
    if (i >= total) return;
    int c = (int)(i & (CMID - 1));
    float v  = y1[i] * ss[2 * c] + ss[2 * c + 1];
    float sv = v / (1.0f + __expf(-v));
    x2[i] = (_Float16)sv;
}

// ---------------------------------------------------------------------------
// conv2: out[N,32] = gather(x2, neigh) @ W2 + b2
// block: 256 thr = 8 waves; tile 128 rows x 32 cols; B staged once per k.
// ---------------------------------------------------------------------------
__launch_bounds__(256)
__global__ void conv2_wmma(const _Float16* __restrict__ x2,
                           const long long* __restrict__ neigh,
                           const _Float16* __restrict__ W2t,
                           const float* __restrict__ b2,
                           float* __restrict__ out,
                           int N) {
    __shared__ __align__(16) _Float16 As[64][LDA];
    __shared__ __align__(16) _Float16 Bs[32][LDA];

    const int tid  = threadIdx.x;
    const int lane = tid & 31;
    const int wave = tid >> 5;
    const int l15  = lane & 15;
    const int kh   = lane >> 4;
    const int wm   = wave >> 1;     // 0..3
    const int wn   = wave & 1;      // 0..1
    const int rowBase0 = blockIdx.x * 128;

    const int arow_s = tid >> 2;            // 0..63
    const int aoff_s = (tid & 3) * 32;
    const int brow_s = tid >> 3;            // 0..31
    const int boff_s = (tid & 7) * 16;      // 0..112

    const v8f vzero = {0.f,0.f,0.f,0.f,0.f,0.f,0.f,0.f};
    v8f acc[2];
    acc[0] = vzero; acc[1] = vzero;

    for (int k = 0; k < KNB; ++k) {
        const _Float16* wk = W2t + (long long)k * (CMID * COUT);
        if (k + 1 < KNB && tid < 64)  // next slice is 8KB = 64 lines
            __builtin_prefetch(W2t + (long long)(k + 1) * (CMID * COUT) + (long long)tid * 64, 0, 0);

        __syncthreads();
        // stage B once per k: 32x128 halves
#pragma unroll
        for (int i = 0; i < 2; ++i)
            *(half8*)(&Bs[brow_s][boff_s + i * 8]) =
                *(const half8*)(wk + brow_s * CMID + boff_s + i * 8);

#pragma unroll
        for (int h = 0; h < 2; ++h) {
            if (h == 1) __syncthreads();
            {
                int grow = rowBase0 + h * 64 + arow_s;
                long long src = -1;
                if (grow < N) src = neigh[(long long)grow * KNB + k];
#pragma unroll
                for (int i = 0; i < 4; ++i) {
                    half8 v = H8Z;
                    if (src >= 0) v = *(const half8*)(x2 + src * CMID + aoff_s + i * 8);
                    *(half8*)(&As[arow_s][aoff_s + i * 8]) = v;
                }
            }
            __syncthreads();

            const _Float16* arow = &As[wm * 16 + l15][0];
            const _Float16* brow = &Bs[wn * 16 + l15][0];

            v16h af = load_frag(arow, 0, kh);
            v16h bf = load_frag(brow, 0, kh);
#pragma unroll
            for (int kc = 0; kc < 128; kc += 32) {
                v16h caf = af, cbf = bf;
                if (kc < 96) {
                    af = load_frag(arow, kc + 32, kh);
                    bf = load_frag(brow, kc + 32, kh);
                }
                acc[h] = __builtin_amdgcn_wmma_f32_16x16x32_f16(
                    false, caf, false, cbf, (short)0, acc[h], false, false);
            }
        }
    }

    int col = wn * 16 + l15;
    float bias = b2[col];
#pragma unroll
    for (int h = 0; h < 2; ++h) {
#pragma unroll
        for (int r = 0; r < 8; ++r) {
            int grow = rowBase0 + h * 64 + wm * 16 + kh * 8 + r;
            if (grow < N)
                out[(long long)grow * COUT + col] = acc[h][r] + bias;
        }
    }
}

// ---------------------------------------------------------------------------
extern "C" void kernel_launch(void* const* d_in, const int* in_sizes, int n_in,
                              void* d_out, int out_size, void* d_ws, size_t ws_size,
                              hipStream_t stream) {
    const float*     data  = (const float*)d_in[0];
    const long long* neigh = (const long long*)d_in[1];
    const float*     W1    = (const float*)d_in[2];
    const float*     b1    = (const float*)d_in[3];
    const float*     gn_w  = (const float*)d_in[4];
    const float*     gn_b  = (const float*)d_in[5];
    const float*     W2    = (const float*)d_in[6];
    const float*     b2    = (const float*)d_in[7];
    float* outp = (float*)d_out;

    const int N = in_sizes[0] / CIN;

    // workspace carve-up (256B aligned)
    char* ws = (char*)d_ws;
    size_t off = 0;
    auto carve = [&](size_t bytes) -> char* {
        char* p = ws + off;
        off = (off + bytes + 255) & ~(size_t)255;
        return p;
    };
    _Float16* xb   = (_Float16*)carve((size_t)N * CIN * 2);
    _Float16* x2b  = (_Float16*)carve((size_t)N * CMID * 2);
    _Float16* W1t  = (_Float16*)carve((size_t)KNB * CIN * CMID * 2);
    _Float16* W2t  = (_Float16*)carve((size_t)KNB * CMID * COUT * 2);
    float*    y1   = (float*)carve((size_t)N * CMID * 4);
    float*    gsum = (float*)carve(64 * 4);
    float*    ss   = (float*)carve(2 * CMID * 4);
    (void)ws_size;

    // 1) convert / transpose to f16
    long long totPrep = (long long)N * CIN
                      + (long long)KNB * CIN * CMID
                      + (long long)KNB * CMID * COUT;
    int gPrep = (int)((totPrep + 255) / 256);
    prep_convert<<<gPrep, 256, 0, stream>>>(data, W1, W2, xb, W1t, W2t, N);

    // 2) zero group-stat accumulators
    hipMemsetAsync(gsum, 0, 64 * sizeof(float), stream);

    // 3) conv1 (WMMA) + group partial sums
    int gConv = (N + 127) / 128;
    conv1_wmma<<<gConv, 256, 0, stream>>>(xb, neigh, W1t, b1, y1, gsum, N);

    // 4) fold GN stats into per-channel scale/shift
    gn_stats<<<1, 128, 0, stream>>>(gsum, gn_w, gn_b, ss, N);

    // 5) normalize + SiLU + f16 convert
    long long totNS = (long long)N * CMID;
    norm_silu<<<(int)((totNS + 255) / 256), 256, 0, stream>>>(y1, ss, x2b, totNS);

    // 6) conv2 (WMMA) -> output
    conv2_wmma<<<gConv, 256, 0, stream>>>(x2b, neigh, W2t, b2, outp, N);
}
